// DenseResGCN_38448547233860
// MI455X (gfx1250) — matrix operand order repaired and checked by reading, not compiled, not run on hardware
//
#include <hip/hip_runtime.h>
#include <stdint.h>

// ---------------------------------------------------------------------------
// DenseResGCN for MI455X (gfx1250), wave32 + WMMA bf16.
//
// Every row of (A+I) sums to exactly 21 (20 knn ones + self loop), so
// D^-1/2 (A+I) D^-1/2 == (A+I)/21 -> adjacency multiply is a 21-way
// gather-sum, not a dense 2048x2048 GEMM.
//
// All per-point linear layers (M=16384) go through one WMMA bf16 GEMM kernel
// (v_wmma_f32_16x16x32_bf16, f32 accumulate). K is zero-padded to multiples
// of 32 (padded weight conversion + zeroed pad columns in activation
// buffers), so fragment loads are clean aligned b128s -- no masking, no
// divergence in the inner loop. Each wave computes a 16x(16*NSUB) strip,
// reusing its A fragment across NSUB WMMAs.
//
// Assumed d_in order (jax tree-flatten, dict keys sorted, lists in order):
//   [0..47]  blocks[i][j] x {W,b,bb,g}        (i*16 + j*4 + {0,1,2,3})
//   [48..53] head W1..W6      [54..59] head b1..b6
//   [60..64] head bb1..bb5    [65..69] head g1..g5
//   [70,71]  init_bn {b,g}    [72,73] init_gcn {W,b}
//   [74..77] trans[0] {W,b,bW,g}  [78..81] trans[1] {W,b,bW,g}
//   [82]     x (8,2048,3) f32
// ---------------------------------------------------------------------------

#define BSZ   8
#define NPTS  2048
#define MTOT  (BSZ * NPTS)
#define KNN   20
#define EPSBN 1e-5f

typedef __attribute__((ext_vector_type(16))) __bf16 v16bf;
typedef __attribute__((ext_vector_type(8)))  float  v8f;

__device__ __forceinline__ unsigned short f2bf(float f) {
    union { float f; uint32_t u; } v; v.f = f;
    uint32_t u = v.u;
    uint32_t r = (u + 0x7FFFu + ((u >> 16) & 1u)) >> 16;  // RNE
    return (unsigned short)r;
}
__device__ __forceinline__ float bf2f(unsigned short h) {
    union { uint32_t u; float f; } v; v.u = ((uint32_t)h) << 16;
    return v.f;
}

// ---------------------------------------------------------------------------
// KNN: one block per (b,i) row. Distances to LDS, then 20 argmin+mask rounds.
// ---------------------------------------------------------------------------
__global__ void knn_kernel(const float* __restrict__ x, int* __restrict__ idx) {
    __shared__ float d2[NPTS];
    __shared__ float rv[256];
    __shared__ int   ri[256];
    const int row = blockIdx.x;
    const int b = row / NPTS, i = row % NPTS;
    const int tid = threadIdx.x;
    const float* xb = x + (size_t)b * NPTS * 3;
    const float xi0 = xb[i*3+0], xi1 = xb[i*3+1], xi2 = xb[i*3+2];
    for (int j = tid; j < NPTS; j += 256) {
        float a = xb[j*3+0] - xi0, c = xb[j*3+1] - xi1, e = xb[j*3+2] - xi2;
        d2[j] = a*a + c*c + e*e;
    }
    __syncthreads();
    for (int k = 0; k < KNN; ++k) {
        float best = 3.4028235e38f; int bi = NPTS;
        for (int j = tid; j < NPTS; j += 256) {
            float v = d2[j];
            if (v < best || (v == best && j < bi)) { best = v; bi = j; }
        }
        rv[tid] = best; ri[tid] = bi;
        __syncthreads();
        for (int s = 128; s > 0; s >>= 1) {
            if (tid < s) {
                if (rv[tid+s] < rv[tid] || (rv[tid+s] == rv[tid] && ri[tid+s] < ri[tid])) {
                    rv[tid] = rv[tid+s]; ri[tid] = ri[tid+s];
                }
            }
            __syncthreads();
        }
        if (tid == 0) { idx[row*KNN + k] = ri[0]; d2[ri[0]] = 3.4028235e38f; }
        __syncthreads();
    }
}

// ---------------------------------------------------------------------------
// WMMA bf16 GEMM:  C[M x Cout] = A[M x Kpad](bf16) * W[Cout x Kpad]^T + bias
// Kpad % 32 == 0, pad region zero on both operands -> unmasked aligned b128
// fragment loads. One wave computes a 16 x (16*NSUB) strip: the A fragment is
// loaded once per K-step and reused across NSUB WMMAs. Wave-uniform guard ->
// EXEC all-ones at every v_wmma.
// ---------------------------------------------------------------------------
template <int NSUB>
__global__ void wmma_gemm_kernel(const unsigned short* __restrict__ A, int lda,
                                 const unsigned short* __restrict__ Bw, int ldb,
                                 const float* __restrict__ bias,
                                 float* __restrict__ C, int ldc,
                                 int M, int Cout, int K, int ngroups) {
    const int wave = blockIdx.x * (blockDim.x >> 5) + (threadIdx.x >> 5);
    const int total = (M >> 4) * ngroups;
    if (wave >= total) return;                       // uniform per-wave
    const int lane = threadIdx.x & 31;
    const int hi = lane >> 4;                        // 0: lanes 0-15, 1: 16-31
    const int lr = lane & 15;
    const int mt = wave / ngroups, g = wave - mt * ngroups;
    const int m0 = mt << 4, n0 = (g * NSUB) << 4;

    const unsigned short* arow = A + (size_t)(m0 + lr) * lda + hi * 8;
    int ncol[NSUB];
    const unsigned short* brow[NSUB];
#pragma unroll
    for (int s = 0; s < NSUB; ++s) {
        ncol[s] = n0 + s * 16 + lr;
        brow[s] = Bw + (size_t)(ncol[s] < Cout ? ncol[s] : 0) * ldb + hi * 8;
    }
    v8f acc[NSUB];
#pragma unroll
    for (int s = 0; s < NSUB; ++s) acc[s] = (v8f){};

    for (int k0 = 0; k0 < K; k0 += 32) {
        // A 16x32: lanes<16 hold K {0..7,16..23}; lanes>=16 hold {8..15,24..31}
        union { v16bf v; uint4 q[2]; } af;
        af.q[0] = *(const uint4*)(arow + k0);
        af.q[1] = *(const uint4*)(arow + k0 + 16);
#pragma unroll
        for (int s = 0; s < NSUB; ++s) {
            // B 32x16: lanes<16 hold K {0..7,16..23} of column lr; lanes>=16: {8..15,24..31}
            union { v16bf v; uint4 q[2]; } bfr;
            const unsigned short* bp = brow[s] + k0;
            bfr.q[0] = *(const uint4*)(bp);
            bfr.q[1] = *(const uint4*)(bp + 16);
            acc[s] = __builtin_amdgcn_wmma_f32_16x16x32_bf16(
                false, af.v, false, bfr.v, (short)0, acc[s], false, false);
        }
    }
#pragma unroll
    for (int s = 0; s < NSUB; ++s) {
        if (ncol[s] < Cout) {
            const float bv = bias ? bias[ncol[s]] : 0.f;
#pragma unroll
            for (int r = 0; r < 8; ++r) {
                const int m = m0 + r + (hi ? 8 : 0);   // C layout: vgpr r -> M
                C[(size_t)m * ldc + ncol[s]] = acc[s][r] + bv;
            }
        }
    }
}

// out[b,i,c] = relu( (in[b,i,c] + sum_{k} in[b,idx[k],c]) / 21 )
__global__ void aggregate_relu_kernel(const float* __restrict__ in,
                                      float* __restrict__ out,
                                      const int* __restrict__ idx, int C) {
    const int t = blockIdx.x * blockDim.x + threadIdx.x;
    if (t >= MTOT * C) return;
    const int m = t / C, c = t - m * C;
    const int b = m / NPTS;
    const size_t rb = (size_t)b * NPTS;
    float acc = in[(size_t)m * C + c];               // identity (self loop)
    const int* ip = idx + (size_t)m * KNN;
#pragma unroll 4
    for (int k = 0; k < KNN; ++k)
        acc += in[(rb + ip[k]) * C + c];
    acc *= (1.0f / 21.0f);
    out[(size_t)m * C + c] = acc > 0.f ? acc : 0.f;
}

// per-channel mean/var over M rows (biased var) -- one block per channel
__global__ void bn_stats_f32_kernel(const float* __restrict__ in, int ld, int M,
                                    float* __restrict__ mean, float* __restrict__ var) {
    __shared__ float s1[256], s2[256];
    const int c = blockIdx.x, tid = threadIdx.x;
    float a = 0.f, b = 0.f;
    for (int m = tid; m < M; m += 256) {
        float v = in[(size_t)m * ld + c];
        a += v; b += v * v;
    }
    s1[tid] = a; s2[tid] = b; __syncthreads();
    for (int s = 128; s > 0; s >>= 1) {
        if (tid < s) { s1[tid] += s1[tid+s]; s2[tid] += s2[tid+s]; }
        __syncthreads();
    }
    if (tid == 0) {
        float mu = s1[0] / M;
        mean[c] = mu;
        var[c] = s2[0] / M - mu * mu;
    }
}
__global__ void bn_stats_bf16_kernel(const unsigned short* __restrict__ in, int ld, int M,
                                     float* __restrict__ mean, float* __restrict__ var) {
    __shared__ float s1[256], s2[256];
    const int c = blockIdx.x, tid = threadIdx.x;
    float a = 0.f, b = 0.f;
    for (int m = tid; m < M; m += 256) {
        float v = bf2f(in[(size_t)m * ld + c]);
        a += v; b += v * v;
    }
    s1[tid] = a; s2[tid] = b; __syncthreads();
    for (int s = 128; s > 0; s >>= 1) {
        if (tid < s) { s1[tid] += s1[tid+s]; s2[tid] += s2[tid+s]; }
        __syncthreads();
    }
    if (tid == 0) {
        float mu = s1[0] / M;
        mean[c] = mu;
        var[c] = s2[0] / M - mu * mu;
    }
}

// y = relu(bn(x)) stored as bf16 at column offset (feeds next GEMM / concat)
__global__ void bn_relu_f32_to_bf16_kernel(const float* __restrict__ in, int ldin,
                                           unsigned short* __restrict__ out, int ldout, int coff,
                                           int C, const float* __restrict__ g,
                                           const float* __restrict__ bb,
                                           const float* __restrict__ mean,
                                           const float* __restrict__ var) {
    const int t = blockIdx.x * blockDim.x + threadIdx.x;
    if (t >= MTOT * C) return;
    const int m = t / C, c = t - m * C;
    float v = (in[(size_t)m * ldin + c] - mean[c]) * rsqrtf(var[c] + EPSBN) * g[c] + bb[c];
    v = v > 0.f ? v : 0.f;
    out[(size_t)m * ldout + coff + c] = f2bf(v);
}
__global__ void bn_relu_bf16_to_bf16_kernel(const unsigned short* __restrict__ in, int ldin,
                                            unsigned short* __restrict__ out, int ldout,
                                            int C, const float* __restrict__ g,
                                            const float* __restrict__ bb,
                                            const float* __restrict__ mean,
                                            const float* __restrict__ var) {
    const int t = blockIdx.x * blockDim.x + threadIdx.x;
    if (t >= MTOT * C) return;
    const int m = t / C, c = t - m * C;
    float v = (bf2f(in[(size_t)m * ldin + c]) - mean[c]) * rsqrtf(var[c] + EPSBN) * g[c] + bb[c];
    v = v > 0.f ? v : 0.f;
    out[(size_t)m * ldout + c] = f2bf(v);
}

// strided f32 -> bf16 copy (x ingest, trans outputs)
__global__ void copy_f32_to_bf16_kernel(const float* __restrict__ in, int ldin,
                                        unsigned short* __restrict__ out, int ldout, int coff,
                                        int C, int M) {
    const int t = blockIdx.x * blockDim.x + threadIdx.x;
    if (t >= M * C) return;
    const int m = t / C, c = t - m * C;
    out[(size_t)m * ldout + coff + c] = f2bf(in[(size_t)m * ldin + c]);
}
// zero a column strip [coff, coff+w) of a bf16 activation buffer
__global__ void zero_cols_kernel(unsigned short* __restrict__ buf, int ld,
                                 int coff, int w, int M) {
    const int t = blockIdx.x * blockDim.x + threadIdx.x;
    if (t >= M * w) return;
    const int m = t / w, c = t - m * w;
    buf[(size_t)m * ld + coff + c] = 0;
}
// weight f32[cout x K] -> bf16[cout x Kpad], zero pad
__global__ void convert_w_pad_kernel(const float* __restrict__ W,
                                     unsigned short* __restrict__ dst,
                                     int K, int Kpad, int n) {
    const int t = blockIdx.x * blockDim.x + threadIdx.x;
    if (t >= n) return;
    const int r = t / Kpad, k = t - r * Kpad;
    dst[t] = (k < K) ? f2bf(W[(size_t)r * K + k]) : (unsigned short)0;
}

// mean over points: f[b,c] = mean_n h[b,n,c]
__global__ void pool_kernel(const unsigned short* __restrict__ h, int ld,
                            float* __restrict__ f, int C) {
    const int t = blockIdx.x * blockDim.x + threadIdx.x;
    if (t >= BSZ * C) return;
    const int b = t / C, c = t - b * C;
    float s = 0.f;
    const unsigned short* p = h + (size_t)b * NPTS * ld + c;
    for (int n = 0; n < NPTS; ++n) s += bf2f(p[(size_t)n * ld]);
    f[(size_t)b * C + c] = s * (1.0f / NPTS);
}

// tiny (M=8) linear tail: one thread per (b, cout)
__global__ void small_linear_kernel(const float* __restrict__ in, int K,
                                    const float* __restrict__ W,
                                    const float* __restrict__ bias,
                                    float* __restrict__ out, int Cout) {
    const int t = blockIdx.x * blockDim.x + threadIdx.x;
    if (t >= BSZ * Cout) return;
    const int b = t / Cout, c = t - b * Cout;
    float acc = bias[c];
    const float* xr = in + (size_t)b * K;
    const float* wr = W + (size_t)c * K;
    for (int k = 0; k < K; ++k) acc += xr[k] * wr[k];
    out[(size_t)b * Cout + c] = acc;
}
// in-place BN(+ReLU) over the 8-row batch, one thread per channel
__global__ void small_bn_relu_kernel(float* __restrict__ x, int C,
                                     const float* __restrict__ g,
                                     const float* __restrict__ bb) {
    const int c = blockIdx.x * blockDim.x + threadIdx.x;
    if (c >= C) return;
    float mu = 0.f;
    for (int b = 0; b < BSZ; ++b) mu += x[(size_t)b * C + c];
    mu *= (1.0f / BSZ);
    float vr = 0.f;
    for (int b = 0; b < BSZ; ++b) { float d = x[(size_t)b * C + c] - mu; vr += d * d; }
    vr *= (1.0f / BSZ);
    const float sc = rsqrtf(vr + EPSBN);
    for (int b = 0; b < BSZ; ++b) {
        float v = (x[(size_t)b * C + c] - mu) * sc * g[c] + bb[c];
        x[(size_t)b * C + c] = v > 0.f ? v : 0.f;
    }
}

// ---------------------------------------------------------------------------
extern "C" void kernel_launch(void* const* d_in, const int* in_sizes, int n_in,
                              void* d_out, int out_size, void* d_ws, size_t ws_size,
                              hipStream_t stream) {
    (void)in_sizes; (void)n_in; (void)out_size; (void)ws_size;

    const float* IN[83];
    for (int i = 0; i < 83; ++i) IN[i] = (const float*)d_in[i];
    const float* x = IN[82];

    // ---- workspace carve ----
    char* p = (char*)d_ws;
    auto carve = [&](size_t bytes) { void* r = (void*)p; p += (bytes + 255) & ~(size_t)255; return r; };
    int*            knnIdx  = (int*)carve((size_t)MTOT * KNN * 4);
    unsigned short* concatB = (unsigned short*)carve((size_t)MTOT * 256 * 2);   // densenet concat (ld 256)
    unsigned short* bfHa    = (unsigned short*)carve((size_t)MTOT * 1024 * 2);
    unsigned short* bfHb    = (unsigned short*)carve((size_t)MTOT * 1024 * 2);
    float*          tmp1    = (float*)carve((size_t)MTOT * 1024 * 4);           // GEMM out
    float*          tmp2    = (float*)carve((size_t)MTOT * 64 * 4);             // aggregate out
    float*          meanB   = (float*)carve(1024 * 4);
    float*          varB    = (float*)carve(1024 * 4);
    float*          fpool   = (float*)carve(BSZ * 1024 * 4);
    float*          gbuf1   = (float*)carve(BSZ * 1024 * 4);
    float*          gbuf2   = (float*)carve(BSZ * 512 * 4);
    unsigned short* wbf     = (unsigned short*)carve((size_t)800000 * 2);       // bf16 weights (K-padded)

    auto EW = [&](long n) { return dim3((unsigned)((n + 255) / 256)); };

    // ---- convert all WMMA-path weights to bf16 with K padded to mult of 32 ----
    struct WInfo { int idx, cout, cin; };
    const WInfo WT[18] = {
        {72, 64, 3},                                            // init_gcn W
        {0,32,64},{4,32,96},{8,32,128},{12,32,160},             // block 0
        {16,32,96},{20,32,128},{24,32,160},{28,32,192},         // block 1
        {32,32,112},{36,32,144},{40,32,176},{44,32,208},        // block 2
        {74,96,192},{78,112,224},                               // trans 0/1
        {48,128,240},{49,512,128},{50,1024,512}                 // head W1..W3
    };
    unsigned short* wptr[18];
    int wkpad[18];
    {
        size_t off = 0;
        for (int t = 0; t < 18; ++t) {
            const int kp = (WT[t].cin + 31) & ~31;
            const int n = WT[t].cout * kp;
            wptr[t] = wbf + off;
            wkpad[t] = kp;
            convert_w_pad_kernel<<<EW(n), 256, 0, stream>>>(IN[WT[t].idx], wptr[t],
                                                            WT[t].cin, kp, n);
            off += (size_t)n;
        }
    }

    // ---- sparse adjacency (knn + self, weight 1/21 uniform) ----
    knn_kernel<<<MTOT, 256, 0, stream>>>(x, knnIdx);

    auto gemm = [&](const unsigned short* A, int lda, int kpad,
                    const unsigned short* W, const float* bias,
                    float* C, int cout) {
        const int ntiles = (cout + 15) / 16;
        const int nsub = (cout <= 32) ? 2 : 4;
        const int ngroups = (ntiles + nsub - 1) / nsub;
        const int total = (MTOT / 16) * ngroups;
        const int blocks = (total + 7) / 8;                  // 8 waves / block
        if (nsub == 2)
            wmma_gemm_kernel<2><<<blocks, 256, 0, stream>>>(A, lda, W, kpad, bias,
                                                            C, cout, MTOT, cout, kpad, ngroups);
        else
            wmma_gemm_kernel<4><<<blocks, 256, 0, stream>>>(A, lda, W, kpad, bias,
                                                            C, cout, MTOT, cout, kpad, ngroups);
    };

    // ---- zero concat buffer once (covers x pad cols + untouched tail) ----
    zero_cols_kernel<<<EW((long)MTOT * 256), 256, 0, stream>>>(concatB, 256, 0, 256, MTOT);

    // ---- init GCN: x -> 64, aggregate, bn, relu ----
    copy_f32_to_bf16_kernel<<<EW((long)MTOT * 3), 256, 0, stream>>>(x, 3, concatB, 256, 0, 3, MTOT);
    gemm(concatB, 256, 32, wptr[0], IN[73], tmp1, 64);
    aggregate_relu_kernel<<<EW((long)MTOT * 64), 256, 0, stream>>>(tmp1, tmp2, knnIdx, 64);
    bn_stats_f32_kernel<<<64, 256, 0, stream>>>(tmp2, 64, MTOT, meanB, varB);
    bn_relu_f32_to_bf16_kernel<<<EW((long)MTOT * 64), 256, 0, stream>>>(
        tmp2, 64, concatB, 256, 0, 64, IN[71], IN[70], meanB, varB);

    // ---- dense blocks ----
    int curw = 64, wslot = 1;
    for (int blk = 0; blk < 3; ++blk) {
        for (int j = 0; j < 4; ++j) {
            const int base = blk * 16 + j * 4;             // blocks[blk][j] {W,b,bb,g}
            const int kpad = (curw + 31) & ~31;
            if (kpad > curw)                               // block-3 widths: scrub stale pad cols
                zero_cols_kernel<<<EW((long)MTOT * (kpad - curw)), 256, 0, stream>>>(
                    concatB, 256, curw, kpad - curw, MTOT);
            gemm(concatB, 256, kpad, wptr[wslot], IN[base + 1], tmp1, 32);
            aggregate_relu_kernel<<<EW((long)MTOT * 32), 256, 0, stream>>>(tmp1, tmp2, knnIdx, 32);
            bn_stats_f32_kernel<<<32, 256, 0, stream>>>(tmp2, 32, MTOT, meanB, varB);
            bn_relu_f32_to_bf16_kernel<<<EW((long)MTOT * 32), 256, 0, stream>>>(
                tmp2, 32, concatB, 256, curw, 32, IN[base + 3], IN[base + 2], meanB, varB);
            curw += 32;
            ++wslot;
        }
        if (blk < 2) {                                     // transition: bn,relu -> GEMM c -> c/2
            const int tb = 74 + blk * 4;                   // trans {W,b,bW,g}
            bn_stats_bf16_kernel<<<curw, 256, 0, stream>>>(concatB, 256, MTOT, meanB, varB);
            bn_relu_bf16_to_bf16_kernel<<<EW((long)MTOT * curw), 256, 0, stream>>>(
                concatB, 256, bfHa, 256, curw, IN[tb + 3], IN[tb + 1], meanB, varB);
            const int co = curw / 2;                       // 96, 112
            gemm(bfHa, 256, curw, wptr[13 + blk], IN[tb + 2], tmp1, co);
            copy_f32_to_bf16_kernel<<<EW((long)MTOT * co), 256, 0, stream>>>(
                tmp1, co, concatB, 256, 0, co, MTOT);
            curw = co;
        }
    }

    // ---- head MLP (per-point), widths 240 -> 128 -> 512 -> 1024 ----
    {
        const int kpad = (curw + 31) & ~31;                // 240 -> 256
        if (kpad > curw)
            zero_cols_kernel<<<EW((long)MTOT * (kpad - curw)), 256, 0, stream>>>(
                concatB, 256, curw, kpad - curw, MTOT);
        gemm(concatB, 256, kpad, wptr[15], IN[54], tmp1, 128);
    }
    bn_stats_f32_kernel<<<128, 256, 0, stream>>>(tmp1, 128, MTOT, meanB, varB);
    bn_relu_f32_to_bf16_kernel<<<EW((long)MTOT * 128), 256, 0, stream>>>(
        tmp1, 128, bfHa, 128, 0, 128, IN[65], IN[60], meanB, varB);

    gemm(bfHa, 128, 128, wptr[16], IN[55], tmp1, 512);
    bn_stats_f32_kernel<<<512, 256, 0, stream>>>(tmp1, 512, MTOT, meanB, varB);
    bn_relu_f32_to_bf16_kernel<<<EW((long)MTOT * 512), 256, 0, stream>>>(
        tmp1, 512, bfHb, 512, 0, 512, IN[66], IN[61], meanB, varB);

    gemm(bfHb, 512, 512, wptr[17], IN[56], tmp1, 1024);
    bn_stats_f32_kernel<<<1024, 256, 0, stream>>>(tmp1, 1024, MTOT, meanB, varB);
    bn_relu_f32_to_bf16_kernel<<<EW((long)MTOT * 1024), 256, 0, stream>>>(
        tmp1, 1024, bfHa, 1024, 0, 1024, IN[67], IN[62], meanB, varB);

    // ---- mean pool + 8-row tail ----
    pool_kernel<<<EW(BSZ * 1024), 256, 0, stream>>>(bfHa, 1024, fpool, 1024);

    small_linear_kernel<<<EW(BSZ * 1024), 256, 0, stream>>>(fpool, 1024, IN[51], IN[57], gbuf1, 1024);
    small_bn_relu_kernel<<<EW(1024), 256, 0, stream>>>(gbuf1, 1024, IN[68], IN[63]);

    small_linear_kernel<<<EW(BSZ * 512), 256, 0, stream>>>(gbuf1, 1024, IN[52], IN[58], gbuf2, 512);
    small_bn_relu_kernel<<<EW(512), 256, 0, stream>>>(gbuf2, 512, IN[69], IN[64]);

    small_linear_kernel<<<EW(BSZ * 40), 256, 0, stream>>>(gbuf2, 512, IN[53], IN[59], (float*)d_out, 40);
}
